// GNN_64295660421514
// MI455X (gfx1250) — matrix operand order repaired and checked by reading, not compiled, not run on hardware
//
#include <hip/hip_runtime.h>

// Problem constants (from reference)
#define N_TOTAL   204800
#define N_EDGES   3276800
#define EMBED     64
#define HID       1600      // 25*64
#define NGRAPH    8192
// Workspace layout (bytes). Requires ws_size >= ~140 MB.
//  [0,           0.82MB)  deg -> dinv
//  [1MB,         54MB)    h0  (fp32 204800x64)   -- later reused as h1 (8192x1600 fp32)
//  [54MB,        107MB)   hagg (fp32 204800x64)
//  [107MB,       134MB)   hagg_bf16 (204800x64)
//  [134MB,       140MB)   W1_bf16 (1600x1600)

typedef __attribute__((ext_vector_type(16))) __bf16 v16bf;
typedef __attribute__((ext_vector_type(8)))  float  v8f;

union ABFrag { v16bf v; unsigned u[8]; unsigned short s[16]; };

__device__ __forceinline__ unsigned short f2bf(float f) {
  unsigned u = __float_as_uint(f);
  u += 0x7FFFu + ((u >> 16) & 1u);          // round-to-nearest-even
  return (unsigned short)(u >> 16);
}
__device__ __forceinline__ unsigned packbf(float lo, float hi) {
  return (unsigned)f2bf(lo) | ((unsigned)f2bf(hi) << 16);
}

// CDNA5 async global->LDS copy of 16 bytes (tracked by ASYNCcnt).
// ldsAddr: wave-relative LDS byte address; gaddr: 64-bit global address.
__device__ __forceinline__ void async_copy_b128(unsigned ldsAddr, unsigned long long gaddr) {
  asm volatile("global_load_async_to_lds_b128 %0, %1, off"
               :: "v"(ldsAddr), "v"(gaddr) : "memory");
}
__device__ __forceinline__ void wait_async0() {
  asm volatile("s_wait_asynccnt 0x0" ::: "memory");
}

// ---------------- 1. init: deg = 1 (self loop), hagg = b_conv ----------------
__global__ void k_init(float* __restrict__ deg, float* __restrict__ hagg,
                       const float* __restrict__ b_conv) {
  size_t i = (size_t)blockIdx.x * blockDim.x + threadIdx.x;  // 0 .. N_TOTAL*64
  hagg[i] = b_conv[i & 63];
  if (i < N_TOTAL) deg[i] = 1.0f;
}

// ---------------- 2. degree accumulate ----------------
__global__ void k_deg(const int* __restrict__ dst, float* __restrict__ deg) {
  size_t e = (size_t)blockIdx.x * blockDim.x + threadIdx.x;
  if (e < N_EDGES) atomicAdd(&deg[dst[e]], 1.0f);
}

// ---------------- 3. dinv = rsqrt(deg), in place ----------------
__global__ void k_rsqrt(float* __restrict__ deg) {
  size_t i = (size_t)blockIdx.x * blockDim.x + threadIdx.x;
  if (i < N_TOTAL) deg[i] = rsqrtf(deg[i]);
}

// ---------------- 4. h0 = x @ W_conv  (bf16 WMMA, fp32 accum) ----------------
// One wave per 16-row strip; computes 16x64 (4 col tiles), K = 64 (2 steps of 32).
__global__ void k_gemm_conv(const float* __restrict__ x, const float* __restrict__ Wc,
                            float* __restrict__ h0) {
  int wid  = (int)((blockIdx.x * blockDim.x + threadIdx.x) >> 5);
  int lane = threadIdx.x & 31;
  if (wid >= N_TOTAL / 16) return;            // wave-uniform: EXEC stays all-ones
  int row0 = wid * 16;
  int m    = lane & 15;
  int hi   = lane >> 4;                        // 0 | 1 (lane half)

  v8f acc[4] = {};
  for (int ks = 0; ks < 2; ++ks) {
    int k0 = ks * 32;
    ABFrag a;
    const float* arow = x + (size_t)(row0 + m) * EMBED + k0;
#pragma unroll
    for (int v = 0; v < 8; ++v) {
      // 16-bit A layout: K = 2v (+8 for v>=4) (+8 for upper lane half)
      int K = 2 * v + ((v >= 4) ? 8 : 0) + (hi ? 8 : 0);
      a.u[v] = packbf(arow[K], arow[K + 1]);
    }
#pragma unroll
    for (int c = 0; c < 4; ++c) {
      ABFrag b;
      int n = c * 16 + m;
#pragma unroll
      for (int v = 0; v < 8; ++v) {
        // 16-bit B layout: K = 2v (+8 for v>=4) (+8 for upper lane half)
        int K = k0 + 2 * v + ((v >= 4) ? 8 : 0) + (hi ? 8 : 0);
        b.u[v] = packbf(Wc[(size_t)K * EMBED + n], Wc[(size_t)(K + 1) * EMBED + n]);
      }
      acc[c] = __builtin_amdgcn_wmma_f32_16x16x32_bf16(
          false, a.v, false, b.v, (short)0, acc[c], false, false);
    }
  }
  int mb = row0 + (hi ? 8 : 0);
#pragma unroll
  for (int c = 0; c < 4; ++c) {
    int n = c * 16 + m;
#pragma unroll
    for (int r = 0; r < 8; ++r)
      h0[(size_t)(mb + r) * EMBED + n] = acc[c][r];   // D: M = r + 8*hi, N = lane%16
  }
}

// ---------------- 5. self-loop contribution (exclusive, no atomics) ----------------
__global__ void k_selfloop(const float* __restrict__ h0, const float* __restrict__ dinv,
                           float* __restrict__ hagg) {
  size_t i = (size_t)blockIdx.x * blockDim.x + threadIdx.x;   // 0 .. N_TOTAL*16
  size_t n = i >> 4;
  size_t off = n * EMBED + (i & 15) * 4;
  float w = dinv[n]; w *= w;
  float4 v = *(const float4*)(h0 + off);
  float4 o = *(const float4*)(hagg + off);
  o.x += v.x * w; o.y += v.y * w; o.z += v.z * w; o.w += v.w * w;
  *(float4*)(hagg + off) = o;
}

// ---------------- 6. edge scatter: wave per edge, lane handles 2 features ----------------
__global__ void k_scatter(const int* __restrict__ src, const int* __restrict__ dst,
                          const float* __restrict__ dinv, const float* __restrict__ h0,
                          float* __restrict__ hagg) {
  size_t t = (size_t)blockIdx.x * blockDim.x + threadIdx.x;
  size_t e = t >> 5;
  int lane = (int)(t & 31);
  if (e >= N_EDGES) return;
  int s = src[e], d = dst[e];
  float w = dinv[s] * dinv[d];
  float2 v = *(const float2*)(h0 + (size_t)s * EMBED + lane * 2);
  float* p = hagg + (size_t)d * EMBED + lane * 2;
  atomicAdd(p,     v.x * w);
  atomicAdd(p + 1, v.y * w);
}

// ---------------- 7. ReLU + convert to bf16 (pack 2) ----------------
__global__ void k_relucvt(const float* __restrict__ hagg, unsigned* __restrict__ hb) {
  size_t i = (size_t)blockIdx.x * blockDim.x + threadIdx.x;   // 0 .. N_TOTAL*32
  float a = fmaxf(hagg[2 * i],     0.0f);
  float b = fmaxf(hagg[2 * i + 1], 0.0f);
  hb[i] = packbf(a, b);
}

// ---------------- 8. W1 -> bf16 ----------------
__global__ void k_cvtW1(const float* __restrict__ W1, unsigned* __restrict__ W1b) {
  size_t i = (size_t)blockIdx.x * blockDim.x + threadIdx.x;   // 0 .. HID*HID/2
  W1b[i] = packbf(W1[2 * i], W1[2 * i + 1]);
}

// ---------------- 9. h1 = relu(hagg @ W1 + b1)  (bf16 WMMA, LDS-staged) --------------
// Block = 8 waves, tile 128M x 64N, K-step 32. Operand tiles staged via
// global_load_async_to_lds_b128 (ASYNCcnt); fragments gathered from LDS.
#define BM 128
#define BN 64
#define BK 32
__global__ void k_gemm_dense(const unsigned short* __restrict__ Ab,
                             const unsigned short* __restrict__ Bb,
                             const float* __restrict__ b1, float* __restrict__ h1) {
  __shared__ unsigned short ldsA[BM * BK];   // 8192 B
  __shared__ unsigned short ldsB[BK * BN];   // 4096 B
  int t    = threadIdx.x;
  int wave = t >> 5, lane = t & 31;
  int wr   = wave >> 2, wc = wave & 3;       // 2 x 4 wave grid over (M, N)
  int bm   = (int)(blockIdx.x % (NGRAPH / BM));   // 64 row blocks
  int bn   = (int)(blockIdx.x / (NGRAPH / BM));   // 25 col blocks
  int row0 = bm * BM, col0 = bn * BN;
  int m    = lane & 15, hi = lane >> 4;

  unsigned ldsA_base = (unsigned)(unsigned long long)(void*)ldsA;
  unsigned ldsB_base = (unsigned)(unsigned long long)(void*)ldsB;

  // Static per-thread copy assignments (16B chunks)
  int ar0 = t >> 2,        as0 = t & 3;        // A chunks 0..255
  int ar1 = (t + 256) >> 2, as1 = t & 3;       // A chunks 256..511
  int br  = t >> 3,        bs  = t & 7;        // B chunks 0..255

  v8f acc[4] = {};
  for (int kb = 0; kb < HID / BK; ++kb) {
    int k0 = kb * BK;
    __syncthreads();                           // LDS safe to overwrite
    // A tile: 128 rows x 64 B
    async_copy_b128(ldsA_base + (unsigned)(ar0 * (BK * 2) + as0 * 16),
                    (unsigned long long)(Ab + (size_t)(row0 + ar0) * HID + k0 + as0 * 8));
    async_copy_b128(ldsA_base + (unsigned)(ar1 * (BK * 2) + as1 * 16),
                    (unsigned long long)(Ab + (size_t)(row0 + ar1) * HID + k0 + as1 * 8));
    // B tile: 32 rows x 128 B
    async_copy_b128(ldsB_base + (unsigned)(br * (BN * 2) + bs * 16),
                    (unsigned long long)(Bb + (size_t)(k0 + br) * HID + col0 + bs * 8));
    wait_async0();
    __syncthreads();                           // tiles visible to all waves

    // B fragment from LDS: lane n = wc*16 + m, K = 2v (+8 for v>=4) (+8 upper half)
    ABFrag b;
    int nl = wc * 16 + m;
#pragma unroll
    for (int v = 0; v < 8; ++v) {
      int K = 2 * v + ((v >= 4) ? 8 : 0) + (hi ? 8 : 0);
      b.s[2 * v]     = ldsB[K * BN + nl];
      b.s[2 * v + 1] = ldsB[(K + 1) * BN + nl];
    }
#pragma unroll
    for (int s4 = 0; s4 < 4; ++s4) {
      // A fragment: row = wr*64 + s4*16 + m, K spans {0-7,16-23} (+8 upper half)
      const unsigned short* ap = ldsA + (wr * 64 + s4 * 16 + m) * BK;
      uint4 lo = *(const uint4*)(ap + hi * 8);        // K 0..7   (+8)
      uint4 hh = *(const uint4*)(ap + 16 + hi * 8);   // K 16..23 (+8)
      ABFrag a;
      a.u[0] = lo.x; a.u[1] = lo.y; a.u[2] = lo.z; a.u[3] = lo.w;
      a.u[4] = hh.x; a.u[5] = hh.y; a.u[6] = hh.z; a.u[7] = hh.w;
      acc[s4] = __builtin_amdgcn_wmma_f32_16x16x32_bf16(
          false, a.v, false, b.v, (short)0, acc[s4], false, false);
    }
  }
  int nn = col0 + wc * 16 + m;
  float bias = b1[nn];
#pragma unroll
  for (int s4 = 0; s4 < 4; ++s4) {
    int mb = row0 + wr * 64 + s4 * 16 + (hi ? 8 : 0);
#pragma unroll
    for (int r = 0; r < 8; ++r)
      h1[(size_t)(mb + r) * HID + nn] = fmaxf(acc[s4][r] + bias, 0.0f);
  }
}

// ---------------- 10. logits + softmax: one wave per graph ----------------
__global__ void k_head(const float* __restrict__ h1, const float* __restrict__ W2,
                       const float* __restrict__ b2, float* __restrict__ out) {
  size_t t = (size_t)blockIdx.x * blockDim.x + threadIdx.x;
  int g = (int)(t >> 5);
  int lane = (int)(t & 31);
  if (g >= NGRAPH) return;
  const float* row = h1 + (size_t)g * HID;
  float s0 = 0.0f, s1 = 0.0f;
  for (int k = lane; k < HID; k += 32) {
    float v = row[k];
    s0 += v * W2[2 * k];
    s1 += v * W2[2 * k + 1];
  }
#pragma unroll
  for (int off = 16; off; off >>= 1) {
    s0 += __shfl_xor(s0, off, 32);
    s1 += __shfl_xor(s1, off, 32);
  }
  if (lane == 0) {
    s0 += b2[0]; s1 += b2[1];
    float mx = fmaxf(s0, s1);
    float e0 = __expf(s0 - mx), e1 = __expf(s1 - mx);
    float inv = 1.0f / (e0 + e1);
    out[2 * g]     = e0 * inv;
    out[2 * g + 1] = e1 * inv;
  }
}

extern "C" void kernel_launch(void* const* d_in, const int* in_sizes, int n_in,
                              void* d_out, int out_size, void* d_ws, size_t ws_size,
                              hipStream_t stream) {
  const float* x   = (const float*)d_in[0];
  const int*   ei  = (const int*)d_in[1];
  /* d_in[2] = batch : unused (graphs are contiguous 25-node blocks) */
  const float* Wc  = (const float*)d_in[3];
  const float* bcv = (const float*)d_in[4];
  const float* W1  = (const float*)d_in[5];
  const float* b1  = (const float*)d_in[6];
  const float* W2  = (const float*)d_in[7];
  const float* b2  = (const float*)d_in[8];
  const int* src = ei;
  const int* dst = ei + N_EDGES;

  char* ws = (char*)d_ws;
  float*    deg  = (float*)ws;
  float*    h0   = (float*)(ws + (size_t)1   * 1024 * 1024);
  float*    hagg = (float*)(ws + (size_t)54  * 1024 * 1024);
  unsigned* hb   = (unsigned*)(ws + (size_t)107 * 1024 * 1024);
  unsigned* W1b  = (unsigned*)(ws + (size_t)134 * 1024 * 1024);
  float*    h1   = h0;   // h0 dead after scatter; reuse 52.4MB region for h1

  k_init      <<<51200,  256, 0, stream>>>(deg, hagg, bcv);
  k_deg       <<<12800,  256, 0, stream>>>(dst, deg);
  k_rsqrt     <<<800,    256, 0, stream>>>(deg);
  k_gemm_conv <<<1600,   256, 0, stream>>>(x, Wc, h0);
  k_selfloop  <<<12800,  256, 0, stream>>>(h0, deg, hagg);
  k_scatter   <<<409600, 256, 0, stream>>>(src, dst, deg, h0, hagg);
  k_relucvt   <<<25600,  256, 0, stream>>>(hagg, hb);
  k_cvtW1     <<<5000,   256, 0, stream>>>(W1, W1b);
  k_gemm_dense<<<1600,   256, 0, stream>>>((const unsigned short*)hb,
                                           (const unsigned short*)W1b, b1, h1);
  k_head      <<<1024,   256, 0, stream>>>(h1, W2, b2, (float*)d_out);
}
